// LDGroupedAutoencoder_26310969655515
// MI455X (gfx1250) — matrix operand order repaired and checked by reading, not compiled, not run on hardware
//
#include <hip/hip_runtime.h>
#include <hip/hip_bf16.h>

// Problem constants (from the reference)
#define B_     128
#define INLEN  200000
#define K20    20
#define G_     10000
#define C_     7
#define GC     70000           // G_*C_
#define GCP    70016           // GC padded to multiple of 32 (WMMA K step)
#define LAT    512
#define SPLITS 64              // split-K factor for fc1

typedef __attribute__((ext_vector_type(16))) __bf16 v16bf;
typedef __attribute__((ext_vector_type(8)))  __bf16 v8bf;
typedef __attribute__((ext_vector_type(8)))  float  v8f;
typedef __attribute__((ext_vector_type(4)))  float  v4f;
typedef __attribute__((ext_vector_type(4)))  unsigned int v4u;
typedef __attribute__((ext_vector_type(8)))  int    v8i;
typedef __attribute__((ext_vector_type(4)))  int    v4i;

__device__ __forceinline__ float leaky_(float x) { return x > 0.f ? x : 0.1f * x; }

// ---------------------------------------------------------------------------
// init: zero the fc1 split-K accumulator and the padded tail columns of h
// ---------------------------------------------------------------------------
__global__ void k_init(float* __restrict__ lat_acc, __bf16* __restrict__ h) {
  int t = blockIdx.x * blockDim.x + threadIdx.x;           // 65536 threads
  if (t < B_ * LAT) lat_acc[t] = 0.f;
  if (t < B_ * 16) {
    int b = t >> 4, j = t & 15;
    h[(size_t)b * GCP + GC + j] = (__bf16)0.f;
  }
}

// ---------------------------------------------------------------------------
// encoder: grouped conv (window==kernel, out length 1) + leaky, store bf16
// ---------------------------------------------------------------------------
__global__ void k_encoder(const float* __restrict__ x, const float* __restrict__ ew,
                          const float* __restrict__ eb, __bf16* __restrict__ h) {
  int t = blockIdx.x * 256 + threadIdx.x;                  // B_*G_ threads exactly
  int b = t / G_, g = t % G_;
  const float* xp = x + (size_t)b * INLEN + (size_t)g * K20;
  float xv[K20];
#pragma unroll
  for (int k = 0; k < K20; ++k) xv[k] = xp[k];
  const float* wp = ew + (size_t)g * C_ * K20;
  const float* bp = eb + (size_t)g * C_;
  __bf16* hp = h + (size_t)b * GCP + (size_t)g * C_;
#pragma unroll
  for (int c = 0; c < C_; ++c) {
    float a = bp[c];
#pragma unroll
    for (int k = 0; k < K20; ++k) a = fmaf(xv[k], wp[c * K20 + k], a);
    hp[c] = (__bf16)leaky_(a);
  }
}

// ---------------------------------------------------------------------------
// helpers: build WMMA fragments
// ---------------------------------------------------------------------------
__device__ __forceinline__ v16bf frag_a(const __bf16* A, int k, int khalf) {
  v8bf alo = *(const v8bf*)(A + k + 16 * khalf);
  v8bf ahi = *(const v8bf*)(A + k + 16 * khalf + 8);
  v16bf a;
#pragma unroll
  for (int i = 0; i < 8; ++i) { a[i] = alo[i]; a[8 + i] = ahi[i]; }
  return a;
}

__device__ __forceinline__ v16bf frag_b_cvt(const v4f* q) {
  v4f q0 = q[0], q1 = q[1], q2 = q[2], q3 = q[3];
  v16bf bm;
#pragma unroll
  for (int i = 0; i < 4; ++i) {
    bm[i]      = (__bf16)q0[i];
    bm[4 + i]  = (__bf16)q1[i];
    bm[8 + i]  = (__bf16)q2[i];
    bm[12 + i] = (__bf16)q3[i];
  }
  return bm;
}

// ---------------------------------------------------------------------------
// fc1: latent_acc[128,512] += h[128,GCP]_bf16 @ fc1_w[512,GC]_f32^T
// split-K WMMA, one 16x16 tile per wave, f32 atomics into workspace.
// Main loop is branch-free; only K-step 2187 (last split) needs the tail path.
// ---------------------------------------------------------------------------
__global__ void k_fc1(const __bf16* __restrict__ h, const float* __restrict__ w,
                      float* __restrict__ acc_out) {
  const int lane  = threadIdx.x & 31;
  const int mt    = threadIdx.x >> 5;     // 8 waves -> 8 M tiles (M=128)
  const int nt    = blockIdx.x & 31;      // 32 N tiles (N=512)
  const int sp    = blockIdx.x >> 5;      // split-K index
  const int l15   = lane & 15;
  const int khalf = lane >> 4;
  const int KSTEPS = GCP / 32;            // 2188
  const int KS_FULL = KSTEPS - 1;         // steps [0, 2187) are fully in-bounds
  const int ks0 = (KSTEPS * sp) / SPLITS;
  const int ks1 = (KSTEPS * (sp + 1)) / SPLITS;
  const int kse = ks1 < KS_FULL ? ks1 : KS_FULL;

  const __bf16* A = h + (size_t)(mt * 16 + l15) * GCP;
  const float*  W = w + (size_t)(nt * 16 + l15) * GC;   // stride GC (unpadded)

  v8f acc = {};
  for (int ks = ks0; ks < kse; ++ks) {
    const int k = ks * 32;
    __builtin_prefetch(A + k + 1024, 0, 3);
    __builtin_prefetch(W + k + 512, 0, 3);
    v16bf a  = frag_a(A, k, khalf);
    v16bf bm = frag_b_cvt((const v4f*)(W + k + 16 * khalf));
    acc = __builtin_amdgcn_wmma_f32_16x16x32_bf16(false, a, false, bm,
                                                  (short)0, acc, false, false);
  }
  if (ks1 > KS_FULL) {                    // last split only: K = 69984..70015
    const int k = KS_FULL * 32;
    v16bf a = frag_a(A, k, khalf);        // h is zero-padded to GCP
    v16bf bm;
    if (khalf == 0) {                     // K 69984..69999 valid
      bm = frag_b_cvt((const v4f*)(W + k));
    } else {                              // K 70000..70015 out of range
#pragma unroll
      for (int i = 0; i < 16; ++i) bm[i] = (__bf16)0.f;
    }
    acc = __builtin_amdgcn_wmma_f32_16x16x32_bf16(false, a, false, bm,
                                                  (short)0, acc, false, false);
  }
  // D layout: VGPR i -> row m0 + 8*khalf + i, col n0 + l15
  float* o = acc_out + (size_t)(mt * 16 + 8 * khalf) * LAT + (nt * 16 + l15);
#pragma unroll
  for (int i = 0; i < 8; ++i) atomicAdd(o + (size_t)i * LAT, acc[i]);
}

// ---------------------------------------------------------------------------
// fc1 epilogue: latent = leaky(acc + bias) -> bf16
// ---------------------------------------------------------------------------
__global__ void k_fc1_epi(const float* __restrict__ acc, const float* __restrict__ bias,
                          __bf16* __restrict__ lat) {
  int t = blockIdx.x * 256 + threadIdx.x;                  // B_*LAT threads
  int n = t & (LAT - 1);
  lat[t] = (__bf16)leaky_(acc[t] + bias[n]);
}

// ---------------------------------------------------------------------------
// fc2: d[128,GC] = leaky(latent[128,LAT]_bf16 @ fc2_w[GC,LAT]_f32^T + b), bf16 out
// The 16x512 f32 weight tile per block (contiguous 32KB) is staged into LDS by
// ONE Tensor-Data-Mover op (issued by wave 0), with TDM row padding (+16B per
// 256 dwords) so the 16-lane fragment reads are bank-spread. All 8 waves then
// read fragments from LDS (ds_load_b128) and convert f32->bf16 in registers.
// ---------------------------------------------------------------------------
__global__ void k_fc2(const __bf16* __restrict__ lat, const float* __restrict__ w,
                      const float* __restrict__ bias, __bf16* __restrict__ d) {
  __shared__ float smW[8320];              // 8192 data + 32*4 pad dwords = 33280 B
  const int lane  = threadIdx.x & 31;
  const int mt    = threadIdx.x >> 5;
  const int nt    = blockIdx.x;            // 0..4374
  const int l15   = lane & 15;
  const int khalf = lane >> 4;

  if (threadIdx.x < 32) {                  // wave 0 issues the TDM descriptor
    unsigned long long ga = (unsigned long long)(uintptr_t)(w + (size_t)nt * 16 * LAT);
    unsigned int ldsa = (unsigned int)(uintptr_t)(&smW[0]);
    // D# group 0: count=1 user descriptor, lds_addr, global_addr[56:0], type=2
    v4u g0;
    g0[0] = 1u;
    g0[1] = ldsa;
    g0[2] = (unsigned int)ga;
    g0[3] = (unsigned int)((ga >> 32) & 0x01FFFFFFull) | (2u << 30);
    // D# group 1: data_size=4B(code 2), pad_enable, pad_interval=256dw(code 7),
    // pad_amount=4dw(code 3); 1-D: tensor_dim0=tile_dim0=8192, dim1=1, stride=8192
    v8i g1;
    g1[0] = (int)((2u << 16) | (1u << 20) | (7u << 22) | (3u << 25));
    g1[1] = (int)(8192u << 16);            // tensor_dim0[15:0] at bits 63:48
    g1[2] = (int)(1u << 16);               // tensor_dim1 = 1
    g1[3] = (int)(8192u << 16);            // tile_dim0 = 8192
    g1[4] = 1;                             // tile_dim1 = 1
    g1[5] = 8192;                          // tensor_dim0_stride[31:0]
    g1[6] = (int)(8192u << 16);            // tensor_dim1_stride[15:0]
    g1[7] = 0;
    v4i g2 = {};                           // <=2D tensor: groups 2/3 unused
    v4i g3 = {};
    v8i g4 = {};                           // 6-arg toolchain variant: extra group
    __builtin_amdgcn_tensor_load_to_lds(g0, g1, g2, g3, g4, 0);
    __builtin_amdgcn_s_wait_tensorcnt(0);
  }
  __syncthreads();

  const __bf16* A = lat + (size_t)(mt * 16 + l15) * LAT;

  v8f acc = {};
#pragma unroll
  for (int ks = 0; ks < LAT / 32; ++ks) {
    const int k = ks * 32;
    v16bf a = frag_a(A, k, khalf);
    // W fragment from padded LDS: row l15, cols kk..kk+15
    const int linear = l15 * LAT + (k + 16 * khalf);
    const v4f* q = (const v4f*)((const char*)smW +
                                ((size_t)linear << 2) + (size_t)((linear >> 8) << 4));
    v16bf bm = frag_b_cvt(q);
    acc = __builtin_amdgcn_wmma_f32_16x16x32_bf16(false, a, false, bm,
                                                  (short)0, acc, false, false);
  }
  const int ncol = nt * 16 + l15;
  const float bb = bias[ncol];
  __bf16* o = d + (size_t)(mt * 16 + 8 * khalf) * GC + ncol;
#pragma unroll
  for (int i = 0; i < 8; ++i) o[(size_t)i * GC] = (__bf16)leaky_(acc[i] + bb);
}

// ---------------------------------------------------------------------------
// decoder: grouped ConvTranspose1d (in length 1 -> out length K) + sigmoid
// ---------------------------------------------------------------------------
__global__ void k_decoder(const __bf16* __restrict__ d, const float* __restrict__ dw,
                          const float* __restrict__ db, float* __restrict__ out) {
  int t = blockIdx.x * 256 + threadIdx.x;                  // B_*G_ threads
  int b = t / G_, g = t % G_;
  const __bf16* dp = d + (size_t)b * GC + (size_t)g * C_;
  float dv[C_];
#pragma unroll
  for (int c = 0; c < C_; ++c) dv[c] = (float)dp[c];
  const float* wp = dw + (size_t)g * C_ * K20;
  const float bb = db[g];
  float y[K20];
#pragma unroll
  for (int k = 0; k < K20; ++k) y[k] = bb;
#pragma unroll
  for (int c = 0; c < C_; ++c)
#pragma unroll
    for (int k = 0; k < K20; ++k) y[k] = fmaf(dv[c], wp[c * K20 + k], y[k]);
  float* op = out + (size_t)b * INLEN + (size_t)g * K20;
#pragma unroll
  for (int k = 0; k < K20; ++k) op[k] = 1.f / (1.f + __expf(-y[k]));
}

// ---------------------------------------------------------------------------
extern "C" void kernel_launch(void* const* d_in, const int* in_sizes, int n_in,
                              void* d_out, int out_size, void* d_ws, size_t ws_size,
                              hipStream_t stream) {
  const float* x   = (const float*)d_in[0];
  const float* ew  = (const float*)d_in[1];
  const float* eb  = (const float*)d_in[2];
  const float* f1w = (const float*)d_in[3];
  const float* f1b = (const float*)d_in[4];
  const float* f2w = (const float*)d_in[5];
  const float* f2b = (const float*)d_in[6];
  const float* dw  = (const float*)d_in[7];
  const float* db  = (const float*)d_in[8];
  float* out = (float*)d_out;

  // workspace layout (~36.4 MB total):
  //   [0, 256KB)          : fc1 split-K f32 accumulator  (128*512)
  //   [256KB, 384KB)      : latent bf16                  (128*512)
  //   [512KB, +17.92MB)   : h bf16, padded stride GCP    (128*70016)
  //   then                : d bf16                       (128*70000)
  char* ws = (char*)d_ws;
  float*  lat_acc = (float*)(ws);
  __bf16* latbf   = (__bf16*)(ws + (256 << 10));
  __bf16* h       = (__bf16*)(ws + (512 << 10));
  __bf16* dmat    = (__bf16*)(ws + (512 << 10) + (size_t)B_ * GCP * 2);

  k_init   <<<256,              256, 0, stream>>>(lat_acc, h);
  k_encoder<<<(B_ * G_) / 256,  256, 0, stream>>>(x, ew, eb, h);
  k_fc1    <<<32 * SPLITS,      256, 0, stream>>>(h, f1w, lat_acc);
  k_fc1_epi<<<(B_ * LAT) / 256, 256, 0, stream>>>(lat_acc, f1b, latbf);
  k_fc2    <<<GC / 16,          256, 0, stream>>>(latbf, f2w, f2b, dmat);
  k_decoder<<<(B_ * G_) / 256,  256, 0, stream>>>(dmat, dw, db, out);
}
